// GATv2_70342974374325
// MI455X (gfx1250) — matrix operand order repaired and checked by reading, not compiled, not run on hardware
//
#include <hip/hip_runtime.h>
#include <hip/hip_bf16.h>
#include <stdint.h>

// ---------------------------------------------------------------------------
// GATv2 (3 layers) for MI455X / gfx1250, wave32, WMMA bf16 path.
// ---------------------------------------------------------------------------

#define HEADS 4
#define DIN   128
#define HID   128      // 4 heads * 32 ch (layers 1,2)
#define FINAL 64       // 4 heads * 16 ch (layer 3)
#define NEG_SLOPE 0.2f

typedef __bf16 v16bf __attribute__((ext_vector_type(16)));
typedef float  v8f   __attribute__((ext_vector_type(8)));
typedef unsigned int u32x4 __attribute__((ext_vector_type(4)));

union FragU { u32x4 q[2]; v16bf f; };

// ---- ordered-float encoding so atomicMax(u32) implements float max -------
__device__ inline unsigned encf(float f) {
    unsigned u = __float_as_uint(f);
    return (u & 0x80000000u) ? ~u : (u | 0x80000000u);
}
__device__ inline float decf(unsigned u) {
    u = (u & 0x80000000u) ? (u & 0x7FFFFFFFu) : ~u;
    return __uint_as_float(u);
}

// ---------------------------------------------------------------------------
// Elementwise fp32 -> bf16
// ---------------------------------------------------------------------------
__global__ void f32_to_bf16_kernel(const float* __restrict__ src,
                                   __bf16* __restrict__ dst, long long n) {
    long long i = (long long)blockIdx.x * blockDim.x + threadIdx.x;
    if (i < n) dst[i] = (__bf16)src[i];
}

// ---------------------------------------------------------------------------
// Transpose + convert weights: W is (K x Nout) fp32, Wt is (Nout x K) bf16.
// Makes WMMA B-fragments contiguous per lane.
// ---------------------------------------------------------------------------
__global__ void wt_transpose_kernel(const float* __restrict__ W,
                                    __bf16* __restrict__ Wt, int K, int Nout) {
    int idx = blockIdx.x * blockDim.x + threadIdx.x;
    if (idx >= K * Nout) return;
    int k = idx % K;
    int n = idx / K;
    Wt[(size_t)n * K + k] = (__bf16)W[(size_t)k * Nout + n];
}

// ---------------------------------------------------------------------------
// Fragment loaders per ISA 7.12.2 layouts (wave32).
// A 16x32 bf16: lanes 0-15 -> rows, K halves 0..7 / 16..23; lanes 16-31 get
// K 8..15 / 24..31 of the same rows.
// ---------------------------------------------------------------------------
__device__ inline v16bf load_afrag(const __bf16* row_k, int ako) {
    FragU u;
    u.q[0] = *(const u32x4*)(row_k + ako);
    u.q[1] = *(const u32x4*)(row_k + 16 + ako);
    return u.f;
}
// B 32x16 bf16 from transposed weights: lane -> column, K halves like A.
__device__ inline v16bf load_bfrag(const __bf16* p) {
    FragU u;
    u.q[0] = *(const u32x4*)(p);
    u.q[1] = *(const u32x4*)(p + 16);
    return u.f;
}

// ---------------------------------------------------------------------------
// Dual-output GEMM: C{l,r}[nrows x NT*16] = A[nrows x 128] * B{l,r}^T
// B given transposed: (NT*16) x 128 bf16.  8 waves/block, 16 rows per wave.
// A-fragment shared between both weight matrices.
// ---------------------------------------------------------------------------
template <int NT>
__global__ __launch_bounds__(256) void gemm_dual_wmma(
    const __bf16* __restrict__ A,
    const __bf16* __restrict__ Bl,
    const __bf16* __restrict__ Br,
    float* __restrict__ Cl,
    float* __restrict__ Cr,
    int nrows) {
    constexpr int K = 128;
    const int lane = threadIdx.x & 31;
    const int wave = threadIdx.x >> 5;
    const int rowBase = blockIdx.x * 128 + wave * 16;

    int arow = rowBase + (lane & 15);
    if (arow >= nrows) arow = nrows - 1;      // clamp: keep EXEC uniform
    const int ako  = (lane >> 4) * 8;          // A K-offset within chunk
    const int bcol = lane & 15;                // B column within tile
    const int bko  = (lane >> 4) * 8;          // B K-offset within chunk

    v8f accL[NT], accR[NT];
#pragma unroll
    for (int i = 0; i < NT; ++i) { accL[i] = v8f{}; accR[i] = v8f{}; }

    const __bf16* arow_p = A + (size_t)arow * K;
#pragma unroll
    for (int kb = 0; kb < K; kb += 32) {
        v16bf a = load_afrag(arow_p + kb, ako);
#pragma unroll
        for (int nt = 0; nt < NT; ++nt) {
            const size_t boff = (size_t)(nt * 16 + bcol) * K + kb + bko;
            v16bf bl = load_bfrag(Bl + boff);
            accL[nt] = __builtin_amdgcn_wmma_f32_16x16x32_bf16(
                false, a, false, bl, (short)0, accL[nt], false, false);
            v16bf br = load_bfrag(Br + boff);
            accR[nt] = __builtin_amdgcn_wmma_f32_16x16x32_bf16(
                false, a, false, br, (short)0, accR[nt], false, false);
        }
    }

    // C layout: VGPR v, lanes 0-15 -> M=v, lanes 16-31 -> M=8+v; N = lane%16.
    const int crow0 = rowBase + (lane >> 4) * 8;
    const int ccol0 = lane & 15;
#pragma unroll
    for (int nt = 0; nt < NT; ++nt) {
#pragma unroll
        for (int v = 0; v < 8; ++v) {
            int row = crow0 + v;
            if (row < nrows) {
                Cl[(size_t)row * (NT * 16) + nt * 16 + ccol0] = accL[nt][v];
                Cr[(size_t)row * (NT * 16) + nt * 16 + ccol0] = accR[nt][v];
            }
        }
    }
}

// ---------------------------------------------------------------------------
// Edge pass 1: e = leaky_relu(xl[src]+xr[dst]) . att[h], segment-max into emax
// One thread per (edge, head); self-loop edges are e >= E.  C is compile-time
// so gathers become unrolled b128 loads (L2-resident tables).
// ---------------------------------------------------------------------------
template <int C>
__global__ void edge_logits_kernel(const long long* __restrict__ ei,
                                   long long E, long long ET,
                                   const float* __restrict__ xl,
                                   const float* __restrict__ xr,
                                   const float* __restrict__ att,
                                   float* __restrict__ ebuf,
                                   unsigned* __restrict__ emax) {
    constexpr int W = HEADS * C;
    long long t = (long long)blockIdx.x * blockDim.x + threadIdx.x;
    if (t >= ET * HEADS) return;
    long long e = t >> 2;
    int h = (int)(t & 3);
    long long src, dst;
    if (e < E) { src = ei[e]; dst = ei[E + e]; }
    else       { src = dst = e - E; }

    const float4* pl = (const float4*)(xl + (size_t)src * W + h * C);
    const float4* pr = (const float4*)(xr + (size_t)dst * W + h * C);
    const float4* pa = (const float4*)(att + h * C);
    float acc = 0.f;
#pragma unroll
    for (int q = 0; q < C / 4; ++q) {
        float4 l = pl[q];
        float4 r = pr[q];
        float4 a = pa[q];
        float m0 = l.x + r.x; m0 = (m0 > 0.f) ? m0 : NEG_SLOPE * m0;
        float m1 = l.y + r.y; m1 = (m1 > 0.f) ? m1 : NEG_SLOPE * m1;
        float m2 = l.z + r.z; m2 = (m2 > 0.f) ? m2 : NEG_SLOPE * m2;
        float m3 = l.w + r.w; m3 = (m3 > 0.f) ? m3 : NEG_SLOPE * m3;
        acc += m0 * a.x + m1 * a.y + m2 * a.z + m3 * a.w;
    }
    ebuf[t] = acc;
    atomicMax(&emax[(size_t)dst * HEADS + h], encf(acc));
}

// ---------------------------------------------------------------------------
// Edge pass 2: ex = exp(e - emax[dst]); segment-sum into denom; ebuf <- ex
// ---------------------------------------------------------------------------
__global__ void edge_exp_kernel(const long long* __restrict__ ei,
                                long long E, long long ET,
                                const unsigned* __restrict__ emax,
                                float* __restrict__ ebuf,
                                float* __restrict__ denom) {
    long long t = (long long)blockIdx.x * blockDim.x + threadIdx.x;
    if (t >= ET * HEADS) return;
    long long e = t >> 2;
    int h = (int)(t & 3);
    long long dst = (e < E) ? ei[E + e] : (e - E);
    float mx = decf(emax[(size_t)dst * HEADS + h]);
    float ex = __expf(ebuf[t] - mx);
    ebuf[t] = ex;
    atomicAdd(&denom[(size_t)dst * HEADS + h], ex);
}

// ---------------------------------------------------------------------------
// Edge pass 3: agg[dst] += (ex/denom[dst]) * xl[src]
// ---------------------------------------------------------------------------
template <int C>
__global__ void edge_aggregate_kernel(const long long* __restrict__ ei,
                                      long long E, long long ET,
                                      const float* __restrict__ xl,
                                      const float* __restrict__ ebuf,
                                      const float* __restrict__ denom,
                                      float* __restrict__ agg) {
    constexpr int W = HEADS * C;
    long long t = (long long)blockIdx.x * blockDim.x + threadIdx.x;
    if (t >= ET * HEADS) return;
    long long e = t >> 2;
    int h = (int)(t & 3);
    long long src, dst;
    if (e < E) { src = ei[e]; dst = ei[E + e]; }
    else       { src = dst = e - E; }
    float alpha = ebuf[t] / denom[(size_t)dst * HEADS + h];
    const float4* pl = (const float4*)(xl + (size_t)src * W + h * C);
    float* po = agg + (size_t)dst * W + h * C;
#pragma unroll
    for (int q = 0; q < C / 4; ++q) {
        float4 l = pl[q];
        atomicAdd(po + 4 * q + 0, alpha * l.x);
        atomicAdd(po + 4 * q + 1, alpha * l.y);
        atomicAdd(po + 4 * q + 2, alpha * l.z);
        atomicAdd(po + 4 * q + 3, alpha * l.w);
    }
}

// ---------------------------------------------------------------------------
// Node post (layers 1,2): hbf = bf16(elu(agg + b))
// ---------------------------------------------------------------------------
__global__ void node_post_kernel(const float* __restrict__ agg,
                                 const float* __restrict__ b,
                                 __bf16* __restrict__ hbf,
                                 long long total, int width) {
    long long i = (long long)blockIdx.x * blockDim.x + threadIdx.x;
    if (i >= total) return;
    float v = agg[i] + b[(int)(i % width)];
    v = (v > 0.f) ? v : (__expf(v) - 1.f);
    hbf[i] = (__bf16)v;
}

// ---------------------------------------------------------------------------
// Final: out = log_softmax(agg + b3) over 64 classes per node.
// ---------------------------------------------------------------------------
__global__ void final_logsoftmax_kernel(const float* __restrict__ agg,
                                        const float* __restrict__ b,
                                        float* __restrict__ out, long long N) {
    long long n = (long long)blockIdx.x * blockDim.x + threadIdx.x;
    if (n >= N) return;
    const float* p = agg + (size_t)n * FINAL;
    float mx = -3.4e38f;
#pragma unroll
    for (int j = 0; j < FINAL; ++j) {
        float v = p[j] + b[j];
        mx = fmaxf(mx, v);
    }
    float s = 0.f;
#pragma unroll
    for (int j = 0; j < FINAL; ++j) s += __expf(p[j] + b[j] - mx);
    float lse = mx + __logf(s);
    float* o = out + (size_t)n * FINAL;
#pragma unroll
    for (int j = 0; j < FINAL; ++j) o[j] = (p[j] + b[j]) - lse;
}

// ---------------------------------------------------------------------------
// Host-side orchestration
// ---------------------------------------------------------------------------
static inline unsigned blocks_for(long long n, int bs) {
    return (unsigned)((n + bs - 1) / bs);
}

extern "C" void kernel_launch(void* const* d_in, const int* in_sizes, int n_in,
                              void* d_out, int out_size, void* d_ws, size_t ws_size,
                              hipStream_t stream) {
    (void)n_in; (void)out_size; (void)ws_size;
    const float*      x    = (const float*)d_in[0];
    const long long*  ei   = (const long long*)d_in[1];
    const float* Wl1 = (const float*)d_in[2];
    const float* Wr1 = (const float*)d_in[3];
    const float* att1= (const float*)d_in[4];
    const float* b1  = (const float*)d_in[5];
    const float* Wl2 = (const float*)d_in[6];
    const float* Wr2 = (const float*)d_in[7];
    const float* att2= (const float*)d_in[8];
    const float* b2  = (const float*)d_in[9];
    const float* Wl3 = (const float*)d_in[10];
    const float* Wr3 = (const float*)d_in[11];
    const float* att3= (const float*)d_in[12];
    const float* b3  = (const float*)d_in[13];

    const long long N  = in_sizes[0] / DIN;   // 50000
    const long long E  = in_sizes[1] / 2;     // 800000
    const long long ET = E + N;               // with self loops

    // -------- workspace carve-out (256B aligned) --------
    char* wsp = (char*)d_ws;
    size_t off = 0;
    auto alloc = [&](size_t bytes) -> void* {
        void* p = wsp + off;
        off += (bytes + 255) & ~(size_t)255;
        return p;
    };
    __bf16* hbf  = (__bf16*)alloc((size_t)N * HID * sizeof(__bf16));  // bf16 features
    __bf16* wt1l = (__bf16*)alloc((size_t)HID * DIN * sizeof(__bf16));
    __bf16* wt1r = (__bf16*)alloc((size_t)HID * DIN * sizeof(__bf16));
    __bf16* wt2l = (__bf16*)alloc((size_t)HID * HID * sizeof(__bf16));
    __bf16* wt2r = (__bf16*)alloc((size_t)HID * HID * sizeof(__bf16));
    __bf16* wt3l = (__bf16*)alloc((size_t)FINAL * HID * sizeof(__bf16));
    __bf16* wt3r = (__bf16*)alloc((size_t)FINAL * HID * sizeof(__bf16));
    float*  xl   = (float*)alloc((size_t)N * HID * sizeof(float));
    float*  xr   = (float*)alloc((size_t)N * HID * sizeof(float));
    float*  ebuf = (float*)alloc((size_t)ET * HEADS * sizeof(float));
    unsigned* emax = (unsigned*)alloc((size_t)N * HEADS * sizeof(unsigned));
    float*  denom= (float*)alloc((size_t)N * HEADS * sizeof(float));
    float*  agg  = (float*)alloc((size_t)N * HID * sizeof(float));

    const long long totEH = ET * HEADS;

    // -------- prep: bf16 features + transposed bf16 weights --------
    f32_to_bf16_kernel<<<blocks_for(N * DIN, 256), 256, 0, stream>>>(x, hbf, N * DIN);
    wt_transpose_kernel<<<blocks_for(DIN * HID, 256), 256, 0, stream>>>(Wl1, wt1l, DIN, HID);
    wt_transpose_kernel<<<blocks_for(DIN * HID, 256), 256, 0, stream>>>(Wr1, wt1r, DIN, HID);
    wt_transpose_kernel<<<blocks_for(HID * HID, 256), 256, 0, stream>>>(Wl2, wt2l, HID, HID);
    wt_transpose_kernel<<<blocks_for(HID * HID, 256), 256, 0, stream>>>(Wr2, wt2r, HID, HID);
    wt_transpose_kernel<<<blocks_for(HID * FINAL, 256), 256, 0, stream>>>(Wl3, wt3l, HID, FINAL);
    wt_transpose_kernel<<<blocks_for(HID * FINAL, 256), 256, 0, stream>>>(Wr3, wt3r, HID, FINAL);

    // ================= layer 1 (128 -> 4x32) =================
    gemm_dual_wmma<8><<<blocks_for(N, 128), 256, 0, stream>>>(hbf, wt1l, wt1r, xl, xr, (int)N);
    hipMemsetAsync(emax, 0, (size_t)N * HEADS * sizeof(unsigned), stream);
    hipMemsetAsync(denom, 0, (size_t)N * HEADS * sizeof(float), stream);
    hipMemsetAsync(agg, 0, (size_t)N * HID * sizeof(float), stream);
    edge_logits_kernel<32><<<blocks_for(totEH, 256), 256, 0, stream>>>(
        ei, E, ET, xl, xr, att1, ebuf, emax);
    edge_exp_kernel<<<blocks_for(totEH, 256), 256, 0, stream>>>(ei, E, ET, emax, ebuf, denom);
    edge_aggregate_kernel<32><<<blocks_for(totEH, 256), 256, 0, stream>>>(
        ei, E, ET, xl, ebuf, denom, agg);
    node_post_kernel<<<blocks_for(N * HID, 256), 256, 0, stream>>>(agg, b1, hbf, N * HID, HID);

    // ================= layer 2 (128 -> 4x32) =================
    gemm_dual_wmma<8><<<blocks_for(N, 128), 256, 0, stream>>>(hbf, wt2l, wt2r, xl, xr, (int)N);
    hipMemsetAsync(emax, 0, (size_t)N * HEADS * sizeof(unsigned), stream);
    hipMemsetAsync(denom, 0, (size_t)N * HEADS * sizeof(float), stream);
    hipMemsetAsync(agg, 0, (size_t)N * HID * sizeof(float), stream);
    edge_logits_kernel<32><<<blocks_for(totEH, 256), 256, 0, stream>>>(
        ei, E, ET, xl, xr, att2, ebuf, emax);
    edge_exp_kernel<<<blocks_for(totEH, 256), 256, 0, stream>>>(ei, E, ET, emax, ebuf, denom);
    edge_aggregate_kernel<32><<<blocks_for(totEH, 256), 256, 0, stream>>>(
        ei, E, ET, xl, ebuf, denom, agg);
    node_post_kernel<<<blocks_for(N * HID, 256), 256, 0, stream>>>(agg, b2, hbf, N * HID, HID);

    // ================= layer 3 (128 -> 4x16) + log_softmax =================
    gemm_dual_wmma<4><<<blocks_for(N, 128), 256, 0, stream>>>(hbf, wt3l, wt3r, xl, xr, (int)N);
    hipMemsetAsync(emax, 0, (size_t)N * HEADS * sizeof(unsigned), stream);
    hipMemsetAsync(denom, 0, (size_t)N * HEADS * sizeof(float), stream);
    hipMemsetAsync(agg, 0, (size_t)N * FINAL * sizeof(float), stream);
    edge_logits_kernel<16><<<blocks_for(totEH, 256), 256, 0, stream>>>(
        ei, E, ET, xl, xr, att3, ebuf, emax);
    edge_exp_kernel<<<blocks_for(totEH, 256), 256, 0, stream>>>(ei, E, ET, emax, ebuf, denom);
    edge_aggregate_kernel<16><<<blocks_for(totEH, 256), 256, 0, stream>>>(
        ei, E, ET, xl, ebuf, denom, agg);
    final_logsoftmax_kernel<<<blocks_for(N, 256), 256, 0, stream>>>(agg, b3, (float*)d_out, N);
}
